// SMHA_52544629899784
// MI455X (gfx1250) — matrix operand order repaired and checked by reading, not compiled, not run on hardware
//
#include <hip/hip_runtime.h>
#include <hip/hip_bf16.h>

typedef __bf16 bf16;
typedef __attribute__((ext_vector_type(16))) __bf16 v16bf;
typedef __attribute__((ext_vector_type(8)))  __bf16 v8bf;
typedef __attribute__((ext_vector_type(4)))  __bf16 v4bf;
typedef __attribute__((ext_vector_type(8)))  float  v8f;
typedef __attribute__((ext_vector_type(4)))  float  v4f;
typedef int v4i __attribute__((vector_size(16)));

#define H_   16
#define HD_  64
#define L_   2048
#define B_   2
#define E_   1024
#define S_   2048

#if defined(__has_builtin)
#if __has_builtin(__builtin_amdgcn_global_load_async_to_lds_b128) && \
    __has_builtin(__builtin_amdgcn_s_wait_asynccnt)
#define USE_ASYNC_LDS 1
#endif
#endif
#ifndef USE_ASYNC_LDS
#define USE_ASYNC_LDS 0
#endif

static __device__ __forceinline__ v8f wmma_bf16(v16bf a, v16bf b, v8f c) {
  return __builtin_amdgcn_wmma_f32_16x16x32_bf16(false, a, false, b, (short)0, c,
                                                 false, false);
}

// 16-byte global -> LDS copy; async (ASYNCcnt-tracked DMA) when available.
// ROCm 7.2 declares the builtin with generic v4i* params (probe-verified).
static __device__ __forceinline__ void copy16_g2l(bf16* l, const bf16* g) {
#if USE_ASYNC_LDS
  __builtin_amdgcn_global_load_async_to_lds_b128((v4i*)(void*)g, (v4i*)(void*)l,
                                                 0, 0);
#else
  *(v8bf*)l = *(const v8bf*)g;
#endif
}
static __device__ __forceinline__ void wait_async_le2() {
#if USE_ASYNC_LDS
  __builtin_amdgcn_s_wait_asynccnt(2);   // allow next stage's 2 DMAs in flight
#endif
}
static __device__ __forceinline__ void wait_async_0() {
#if USE_ASYNC_LDS
  __builtin_amdgcn_s_wait_asynccnt(0);
#endif
}

// ---- fragment loaders (contiguous k order K = 16g+0..15, same in A and B) ----------

// A fragment: 16x32 bf16, row-major source, ld elements.
// lane = 16g+m: elems 0..15 -> K = 16g+0..15
static __device__ __forceinline__ v16bf load_a_bf16(const bf16* src, int ld) {
  int lane = threadIdx.x & 31;
  int m = lane & 15, g = lane >> 4;
  const bf16* p = src + m * ld + g * 16;
  return *(const v16bf*)(p);
}

// B fragment: 32x16, column n of B contiguous at src + n*ld (one 32B load).
// lane = 16g+n: elems 0..15 -> K = 16g+0..15 of column n
static __device__ __forceinline__ v16bf load_b_col_bf16(const bf16* src, int ld) {
  int lane = threadIdx.x & 31;
  int n = lane & 15, g = lane >> 4;
  return *(const v16bf*)(src + n * ld + g * 16);
}

// ---- shared GEMM core: 16(M) x 64(N) x 1024(K), ping-pong pipelined -------
// Two disjoint register sets (A/B) alternate between "being loaded" and
// "feeding WMMAs" -> no cross-iteration register copies, loads for one
// double-step are in flight while the other computes.
static __device__ __forceinline__ void gemm_16x64_k1024(
    const bf16* __restrict__ Xrow,   // X + m0*E_
    const bf16* __restrict__ Wcol,   // W + n0*E_ (columns n0.. contiguous)
    v8f acc[4]) {
  v16bf aA = load_a_bf16(Xrow, E_);
  v16bf bA[4];
#pragma unroll
  for (int nt = 0; nt < 4; ++nt)
    bA[nt] = load_b_col_bf16(Wcol + (size_t)(nt * 16) * E_, E_);

  v16bf aB, bB[4];
  for (int d = 0; d < 15; ++d) {       // double-steps over K (chunks of 2x32)
    int k1 = d * 64 + 32;
    aB = load_a_bf16(Xrow + k1, E_);
#pragma unroll
    for (int nt = 0; nt < 4; ++nt)
      bB[nt] = load_b_col_bf16(Wcol + (size_t)(nt * 16) * E_ + k1, E_);
#pragma unroll
    for (int nt = 0; nt < 4; ++nt) acc[nt] = wmma_bf16(aA, bA[nt], acc[nt]);

    int k2 = d * 64 + 64;
    aA = load_a_bf16(Xrow + k2, E_);
#pragma unroll
    for (int nt = 0; nt < 4; ++nt)
      bA[nt] = load_b_col_bf16(Wcol + (size_t)(nt * 16) * E_ + k2, E_);
#pragma unroll
    for (int nt = 0; nt < 4; ++nt) acc[nt] = wmma_bf16(aB, bB[nt], acc[nt]);
  }
  // peeled last double-step (chunks 960 and 992): no trailing reload
  aB = load_a_bf16(Xrow + 992, E_);
#pragma unroll
  for (int nt = 0; nt < 4; ++nt)
    bB[nt] = load_b_col_bf16(Wcol + (size_t)(nt * 16) * E_ + 992, E_);
#pragma unroll
  for (int nt = 0; nt < 4; ++nt) acc[nt] = wmma_bf16(aA, bA[nt], acc[nt]);
#pragma unroll
  for (int nt = 0; nt < 4; ++nt) acc[nt] = wmma_bf16(aB, bB[nt], acc[nt]);
}

// ---- kernel 0: one-shot f32 -> bf16 conversion (vectorized) ---------------
__global__ __launch_bounds__(256) void cvt_kernel(const float* __restrict__ src,
                                                  bf16* __restrict__ dst,
                                                  int n4) {
  int i = blockIdx.x * blockDim.x + threadIdx.x;
  if (i < n4) {
    v4f x = *(const v4f*)(src + 4 * (size_t)i);
    v4bf y;
#pragma unroll
    for (int j = 0; j < 4; ++j) y[j] = (bf16)x[j];
    *(v4bf*)(dst + 4 * (size_t)i) = y;
  }
}

// ---- kernel 1: packed in-projection q/k/v (z = 0/1/2), all-bf16 GEMM ------
__global__ __launch_bounds__(256) void proj_kernel(
    const bf16* __restrict__ xq, const bf16* __restrict__ xk,
    const bf16* __restrict__ xv, const bf16* __restrict__ wb,
    const float* __restrict__ bias, bf16* __restrict__ qs,
    bf16* __restrict__ kk, bf16* __restrict__ vt) {
  int type = blockIdx.z;
  const bf16* X  = (type == 0) ? xq : (type == 1) ? xk : xv;
  const bf16* W  = wb + (size_t)type * E_ * E_;
  const float* Bp = bias + type * E_;

  int wave = threadIdx.x >> 5;
  int lane = threadIdx.x & 31;
  int m0 = (blockIdx.x * 8 + wave) * 16;
  int n0 = blockIdx.y * 64;

  v8f z = {0.f, 0.f, 0.f, 0.f, 0.f, 0.f, 0.f, 0.f};
  v8f acc[4] = {z, z, z, z};
  gemm_16x64_k1024(X + (size_t)m0 * E_, W + (size_t)n0 * E_, acc);

  int g = lane >> 4, nl = lane & 15;
#pragma unroll
  for (int nt = 0; nt < 4; ++nt) {
    int e = n0 + nt * 16 + nl;
    float bv = Bp[e];
    int h = e >> 6, d = e & 63;
#pragma unroll
    for (int r = 0; r < 8; ++r) {
      int t = m0 + 8 * g + r;          // token row = l*B + b
      int bb = t % B_, row = t / B_;
      float val = acc[nt][r] + bv;
      if (type == 0) {
        qs[((size_t)(bb * H_ + h) * L_ + row) * HD_ + d] = (bf16)(val * 0.125f);
      } else if (type == 1) {
        kk[((size_t)(bb * H_ + h) * S_ + row) * HD_ + d] = (bf16)val;
      } else {
        vt[((size_t)(bb * H_ + h) * HD_ + d) * S_ + row] = (bf16)val;
      }
    }
  }
}

// ---- kernel 2: sigmoid attention ------------------------------------------
// One block per (b,h, 128 query rows): 8 waves x 16 rows. K/V tiles for each
// 32-wide S step are DMA'd into LDS once per block (double-buffered,
// ASYNCcnt-tracked) and shared by all 8 waves -> 8x less L2 read traffic.
__global__ __launch_bounds__(256) void attn_kernel(
    const bf16* __restrict__ qs, const bf16* __restrict__ kk,
    const bf16* __restrict__ vt, bf16* __restrict__ attn_out,
    float* __restrict__ avg) {
  __shared__ __align__(64) bf16 ldsK[2][32 * HD_];  // 2 x 4KB (32 s-rows x 64)
  __shared__ __align__(64) bf16 ldsV[2][HD_ * 32];  // 2 x 4KB (64 d-rows x 32)
  __shared__ __align__(64) bf16 probs[8][16 * 32];  // per-wave prob tile

  int wave = threadIdx.x >> 5;
  int lane = threadIdx.x & 31;
  int g = lane >> 4, nl = lane & 15;

  int bh = blockIdx.y;
  int b = bh / H_, h = bh % H_;
  int l0 = (blockIdx.x * 8 + wave) * 16;

  const bf16* qh = qs + (size_t)bh * L_ * HD_;
  const bf16* kh = kk + (size_t)bh * S_ * HD_;
  const bf16* vh = vt + (size_t)bh * HD_ * S_;

  // per-thread 16B staging slice: K tile is one contiguous 4KB block,
  // V tile gathers 64 row-segments of 32 elements (row stride S_).
  int off = threadIdx.x * 8;            // bf16 elements, 8 = 16 bytes
  int vd = off >> 5, vj = off & 31;

  // Q row-tile (16 x 64) stays in registers for the whole S loop.
  v16bf qa0 = load_a_bf16(qh + (size_t)l0 * HD_, HD_);
  v16bf qa1 = load_a_bf16(qh + (size_t)l0 * HD_ + 32, HD_);

  v8f z = {0.f, 0.f, 0.f, 0.f, 0.f, 0.f, 0.f, 0.f};
  v8f o[4] = {z, z, z, z};
  v8f den = z;

  // prologue: stage tile 0
  copy16_g2l(&ldsK[0][off], kh + off);
  copy16_g2l(&ldsV[0][off], vh + (size_t)vd * S_ + vj);

  const int NIT = S_ / 32;
  for (int it = 0; it < NIT; ++it) {
    int s0 = it * 32;
    int buf = it & 1;
    if (it + 1 < NIT) {   // kick off DMA of the next tile into the other buffer
      copy16_g2l(&ldsK[buf ^ 1][off], kh + (size_t)(s0 + 32) * HD_ + off);
      copy16_g2l(&ldsV[buf ^ 1][off], vh + (size_t)vd * S_ + (s0 + 32) + vj);
      wait_async_le2();                 // tile `it` landed; next still in flight
    } else {
      wait_async_0();
    }
    __syncthreads();                    // tile `it` visible to all waves

#pragma unroll
    for (int c = 0; c < 2; ++c) {
      int sb = s0 + c * 16;
      v16bf bk0 = load_b_col_bf16(&ldsK[buf][(c * 16) * HD_], HD_);
      v16bf bk1 = load_b_col_bf16(&ldsK[buf][(c * 16) * HD_ + 32], HD_);
      v8f sc = z;
      sc = wmma_bf16(qa0, bk0, sc);     // K-dim 0..31 of hd
      sc = wmma_bf16(qa1, bk1, sc);     // K-dim 32..63 of hd
#pragma unroll
      for (int r = 0; r < 8; ++r) {
        float p = 1.0f / (1.0f + __expf(-sc[r]));   // sigmoid
        den[r] += p;
        int l = l0 + 8 * g + r;
        __hip_atomic_fetch_add(&avg[((size_t)b * L_ + l) * S_ + sb + nl],
                               p * (1.0f / (float)H_), __ATOMIC_RELAXED,
                               __HIP_MEMORY_SCOPE_AGENT);
        probs[wave][(8 * g + r) * 32 + c * 16 + nl] = (bf16)p;  // D -> A relayout
      }
    }
    // wave-local LDS round-trip gives the A fragment of probabilities
    v16bf pa = load_a_bf16(probs[wave], 32);
#pragma unroll
    for (int dt = 0; dt < 4; ++dt) {
      v16bf bv = load_b_col_bf16(&ldsV[buf][(dt * 16) * 32], 32);
      o[dt] = wmma_bf16(pa, bv, o[dt]);
    }
    __syncthreads();   // everyone done with tile `it` before it is restaged
  }

  // row-sum denominators: butterfly across the 16 lanes of each half-wave
#pragma unroll
  for (int r = 0; r < 8; ++r) {
    float v = den[r];
    v += __shfl_xor(v, 1, 32);
    v += __shfl_xor(v, 2, 32);
    v += __shfl_xor(v, 4, 32);
    v += __shfl_xor(v, 8, 32);
    den[r] = 1.0f / (v + 1e-4f);
  }

#pragma unroll
  for (int dt = 0; dt < 4; ++dt) {
#pragma unroll
    for (int r = 0; r < 8; ++r) {
      int l = l0 + 8 * g + r;
      attn_out[((size_t)l * B_ + b) * E_ + h * HD_ + dt * 16 + nl] =
          (bf16)(o[dt][r] * den[r]);
    }
  }
}

// ---- kernel 3: out-projection (all-bf16) ----------------------------------
__global__ __launch_bounds__(256) void outproj_kernel(
    const bf16* __restrict__ X, const bf16* __restrict__ W,
    const float* __restrict__ Bp, float* __restrict__ out) {
  int wave = threadIdx.x >> 5;
  int lane = threadIdx.x & 31;
  int m0 = (blockIdx.x * 8 + wave) * 16;
  int n0 = blockIdx.y * 64;

  v8f z = {0.f, 0.f, 0.f, 0.f, 0.f, 0.f, 0.f, 0.f};
  v8f acc[4] = {z, z, z, z};
  gemm_16x64_k1024(X + (size_t)m0 * E_, W + (size_t)n0 * E_, acc);

  int g = lane >> 4, nl = lane & 15;
#pragma unroll
  for (int nt = 0; nt < 4; ++nt) {
    int e = n0 + nt * 16 + nl;
    float bv = Bp[e];
#pragma unroll
    for (int r = 0; r < 8; ++r) {
      out[(size_t)(m0 + 8 * g + r) * E_ + e] = acc[nt][r] + bv;
    }
  }
}

extern "C" void kernel_launch(void* const* d_in, const int* in_sizes, int n_in,
                              void* d_out, int out_size, void* d_ws,
                              size_t ws_size, hipStream_t stream) {
  (void)in_sizes; (void)n_in; (void)out_size; (void)ws_size;
  const float* query = (const float*)d_in[0];
  const float* keyp  = (const float*)d_in[1];
  const float* value = (const float*)d_in[2];
  const float* ipw   = (const float*)d_in[3];
  const float* ipb   = (const float*)d_in[4];
  const float* out_w = (const float*)d_in[5];
  const float* out_b = (const float*)d_in[6];

  float* out = (float*)d_out;                       // (L,B,E) f32
  float* avg = out + (size_t)L_ * B_ * E_;          // (B,L,S) f32

  const size_t n_act = (size_t)L_ * B_ * E_;        // 4M elements
  bf16* qs  = (bf16*)d_ws;                          // (B*H, L, 64)
  bf16* kk  = qs  + n_act;                          // (B*H, S, 64)
  bf16* vt  = kk  + n_act;                          // (B*H, 64, S)
  bf16* ao  = vt  + n_act;                          // (L, B, E)
  bf16* xq  = ao  + n_act;                          // bf16 copies of inputs
  bf16* xk  = xq  + n_act;
  bf16* xv  = xk  + n_act;
  bf16* wb  = xv  + n_act;                          // bf16 in_proj_weight (3E,E)
  bf16* owb = wb  + (size_t)3 * E_ * E_;            // bf16 out_w (E,E)

  (void)hipMemsetAsync(avg, 0, (size_t)B_ * L_ * S_ * sizeof(float), stream);

  // one-shot bf16 conversions (elementwise, ~3us of HBM/L2 traffic)
  const int TPB = 256;
  int n4_act = (int)(n_act / 4);
  cvt_kernel<<<(n4_act + TPB - 1) / TPB, TPB, 0, stream>>>(query, xq, n4_act);
  cvt_kernel<<<(n4_act + TPB - 1) / TPB, TPB, 0, stream>>>(keyp, xk, n4_act);
  cvt_kernel<<<(n4_act + TPB - 1) / TPB, TPB, 0, stream>>>(value, xv, n4_act);
  int n4_w = 3 * E_ * E_ / 4;
  cvt_kernel<<<(n4_w + TPB - 1) / TPB, TPB, 0, stream>>>(ipw, wb, n4_w);
  int n4_ow = E_ * E_ / 4;
  cvt_kernel<<<(n4_ow + TPB - 1) / TPB, TPB, 0, stream>>>(out_w, owb, n4_ow);

  // 4096 rows / (8 waves * 16) = 32 blocks in x; 1024 cols / 64 = 16 in y
  proj_kernel<<<dim3(32, 16, 3), 256, 0, stream>>>(xq, xk, xv, wb, ipb,
                                                   qs, kk, vt);
  // 2048 rows / (8 waves * 16) = 16 blocks in x; B*H = 32 in y
  attn_kernel<<<dim3(16, 32), 256, 0, stream>>>(qs, kk, vt, ao, avg);
  outproj_kernel<<<dim3(32, 16), 256, 0, stream>>>(ao, owb, out_b, out);
}
